// Compiled_38663295599268
// MI455X (gfx1250) — matrix operand order repaired and checked by reading, not compiled, not run on hardware
//
#include <hip/hip_runtime.h>
#include <hip/hip_bf16.h>
#include <stdint.h>

// CDNA5 wave32 WMMA types
typedef __attribute__((ext_vector_type(16))) __bf16 v16bf;
typedef __attribute__((ext_vector_type(8)))  float  v8f;

#define N_MODELS 64
#define N_LAYERS 4
#define N_FEAT 256
#define ROWS_PER_MODEL 4096
#define KT 8                 // K tiles of 32 per layer (256/32)
#define NT 16                // N tiles of 16 per layer (256/16)
#define FRAGS_PER_ML (KT*NT) // 128 B-fragments per (model,layer)
#define FRAG_ELEMS 512       // 32x16 bf16 per fragment
#define W_ELEMS ((size_t)N_MODELS*N_LAYERS*N_FEAT*N_FEAT) // 16,777,216

__device__ __forceinline__ unsigned short bf16_rne(float x) {
  unsigned u = __builtin_bit_cast(unsigned, x);
  u += 0x7FFFu + ((u >> 16) & 1u);
  return (unsigned short)(u >> 16);
}

union BF16Frag { v16bf v; unsigned short u[16]; uint4 q[2]; };

// ---------------------------------------------------------------------------
// Kernel 1: split fp32 weights into bf16 hi/lo, stored pre-swizzled in the
// B-matrix (32x16, K-major) per-lane layout used here (contiguous K, not the ISA table order):
//   lane c: column n = j*16 + (c&15); K half = (c>>4)*16; VGPR v packs K pair
//   (2v, 2v+1)  ->  per lane 16 contiguous bf16 (32B) in the fragment.
// One thread per (fragment, lane).
// ---------------------------------------------------------------------------
__global__ void __launch_bounds__(256)
weight_split_kernel(const float* __restrict__ Ws,
                    unsigned short* __restrict__ whi,
                    unsigned short* __restrict__ wlo) {
  int tid  = blockIdx.x * 256 + threadIdx.x;
  int lane = tid & 31;
  int frag = tid >> 5;            // ((m*4+l)*8 + t)*16 + j
  int j  = frag & 15;
  int t  = (frag >> 4) & 7;
  int ml = frag >> 7;             // m*N_LAYERS + l
  const float* src = Ws + (size_t)ml * (N_FEAT * N_FEAT);
  int n     = j * 16 + (lane & 15);
  int kbase = t * 32 + ((lane >> 4) & 1) * 16;

  BF16Frag hi, lo;
#pragma unroll
  for (int e = 0; e < 16; ++e) {
    float w = src[(size_t)(kbase + e) * N_FEAT + n];
    unsigned uh = __builtin_bit_cast(unsigned, w) & 0xFFFF0000u; // truncated hi
    hi.u[e] = (unsigned short)(uh >> 16);
    lo.u[e] = bf16_rne(w - __builtin_bit_cast(float, uh));
  }
  size_t off = (size_t)frag * FRAG_ELEMS + (size_t)lane * 16;
  uint4* dh = (uint4*)(whi + off);
  uint4* dl = (uint4*)(wlo + off);
  dh[0] = hi.q[0]; dh[1] = hi.q[1];
  dl[0] = lo.q[0]; dl[1] = lo.q[1];
}

// ---------------------------------------------------------------------------
// Kernel 2: fused 4-layer grouped GEMM. Each wave owns 16 rows of one model,
// keeps them in its private LDS tile across layers, and accumulates all 256
// output features (16 N-tiles) in f32 WMMA accumulators. B fragments use a
// depth-2 prefetch ring (8 b128 loads in flight); the t-loop is unrolled 2x
// so carried values rename instead of generating mov/hazard-NOP chains.
// ---------------------------------------------------------------------------
__global__ void __launch_bounds__(128)
ensemble_mlp_kernel(const float* __restrict__ x,
                    const float* __restrict__ bs,
                    const unsigned short* __restrict__ whi,
                    const unsigned short* __restrict__ wlo,
                    float* __restrict__ out) {
  extern __shared__ float hbuf[];              // 4 waves * 16*256 f32 = 64 KB
  const int lane      = threadIdx.x & 31;
  const int wave      = threadIdx.x >> 5;      // 0..3
  const int wgRowBase = blockIdx.x * 64;       // 64 rows per block
  const int waveRow   = wgRowBase + wave * 16; // 16 rows per wave
  const int m         = wgRowBase / ROWS_PER_MODEL;
  float* hb = hbuf + (size_t)wave * (16 * N_FEAT);

  // Stage this wave's 16x256 fp32 input tile into LDS (coalesced b128).
  {
    const float4* gx = (const float4*)(x + (size_t)waveRow * N_FEAT);
    float4* lx = (float4*)hb;
#pragma unroll
    for (int i = 0; i < 32; ++i)               // 1024 float4 / 32 lanes
      lx[lane + i * 32] = gx[lane + i * 32];
  }

  const int row  = lane & 15;                  // A-matrix row / D column-lane
  const int half = (lane >> 4) & 1;
  const v8f zero = {0.f, 0.f, 0.f, 0.f, 0.f, 0.f, 0.f, 0.f};
  const float* hrowBase = hb + (size_t)row * N_FEAT + half * 16;

  // Raw-A fetch for K-tile t: one 64B chunk per lane from the LDS tile.
  // A order chosen here (16x32, not the ISA table order): lane<16 -> K {0..15}; lane>=16 ->
  // K {16..31}; vector elems 0..15 contiguous (matches B fragments).
  auto loadA = [&](int t, float4* c) {
    const float* hrow = hrowBase + t * 32;
    c[0] = *(const float4*)(hrow + 0);
    c[1] = *(const float4*)(hrow + 4);
    c[2] = *(const float4*)(hrow + 8);
    c[3] = *(const float4*)(hrow + 12);
  };

#pragma unroll 1
  for (int l = 0; l < N_LAYERS; ++l) {
    const int ml = m * N_LAYERS + l;
    const unsigned short* bh = whi + (size_t)ml * FRAGS_PER_ML * FRAG_ELEMS;
    const unsigned short* bl = wlo + (size_t)ml * FRAGS_PER_ML * FRAG_ELEMS;
    const size_t laneOff = (size_t)lane * 16;

    v8f acc[NT];
#pragma unroll
    for (int j = 0; j < NT; ++j) acc[j] = zero;

    // Prime pipelines: raw A for t=0, B fragments 0 and 1 (depth-2 ring).
    float4 araw[4];
    loadA(0, araw);
    v16bf rbh[2], rbl[2];
    rbh[0] = *(const v16bf*)(bh + laneOff);
    rbl[0] = *(const v16bf*)(bl + laneOff);
    rbh[1] = *(const v16bf*)(bh + FRAG_ELEMS + laneOff);
    rbl[1] = *(const v16bf*)(bl + FRAG_ELEMS + laneOff);

#pragma unroll 2
    for (int t = 0; t < KT; ++t) {
      // Convert prefetched raw A floats to bf16 hi/lo fragments (VALU,
      // co-executes with the XDL WMMA pipe).
      BF16Frag ahi, alo;
      const float* af = (const float*)araw;
#pragma unroll
      for (int i = 0; i < 8; ++i) {
        float w1 = af[i];
        float w2 = af[8 + i];
        unsigned u1 = __builtin_bit_cast(unsigned, w1) & 0xFFFF0000u;
        unsigned u2 = __builtin_bit_cast(unsigned, w2) & 0xFFFF0000u;
        ahi.u[i]     = (unsigned short)(u1 >> 16);
        ahi.u[8 + i] = (unsigned short)(u2 >> 16);
        alo.u[i]     = bf16_rne(w1 - __builtin_bit_cast(float, u1));
        alo.u[8 + i] = bf16_rne(w2 - __builtin_bit_cast(float, u2));
      }

      float4 nraw[4];
#pragma unroll
      for (int j = 0; j < NT; ++j) {
        const int fi   = t * NT + j;
        const int slot = fi & 1;
        // Take the prefetched fragment out of the ring, then immediately
        // refill the slot two tiles ahead (loads retire while fi computes).
        v16bf bhi = rbh[slot];
        v16bf blo = rbl[slot];
        if (fi + 2 < FRAGS_PER_ML) {
          size_t noff = (size_t)(fi + 2) * FRAG_ELEMS + laneOff;
          rbh[slot] = *(const v16bf*)(bh + noff);
          rbl[slot] = *(const v16bf*)(bl + noff);
        }
        if (j == 0 && t + 1 < KT) loadA(t + 1, nraw);  // LDS prefetch

        // bf16x3: hi*hi + lo*hi + hi*lo ~ fp32 accuracy.
        acc[j] = __builtin_amdgcn_wmma_f32_16x16x32_bf16(
            false, ahi.v, false, bhi, (short)0, acc[j], false, false);
        acc[j] = __builtin_amdgcn_wmma_f32_16x16x32_bf16(
            false, alo.v, false, bhi, (short)0, acc[j], false, false);
        acc[j] = __builtin_amdgcn_wmma_f32_16x16x32_bf16(
            false, ahi.v, false, blo, (short)0, acc[j], false, false);
      }
      araw[0] = nraw[0]; araw[1] = nraw[1];
      araw[2] = nraw[2]; araw[3] = nraw[3];
    }

    // Epilogue: bias add, then scatter D layout (row = half*8 + r,
    // col = j*16 + row-lane) back to LDS, or to global for the last layer.
    const float* bsl = bs + (size_t)ml * N_FEAT;
    if (l < N_LAYERS - 1) {
#pragma unroll
      for (int j = 0; j < NT; ++j) {
        float bv = bsl[j * 16 + row];
#pragma unroll
        for (int r = 0; r < 8; ++r)
          hb[(size_t)(half * 8 + r) * N_FEAT + j * 16 + row] = acc[j][r] + bv;
      }
    } else {
      float* orow = out + (size_t)waveRow * N_FEAT;
#pragma unroll
      for (int j = 0; j < NT; ++j) {
        float bv = bsl[j * 16 + row];
#pragma unroll
        for (int r = 0; r < 8; ++r)
          orow[(size_t)(half * 8 + r) * N_FEAT + j * 16 + row] = acc[j][r] + bv;
      }
    }
  }
}

extern "C" void kernel_launch(void* const* d_in, const int* in_sizes, int n_in,
                              void* d_out, int out_size, void* d_ws, size_t ws_size,
                              hipStream_t stream) {
  const float* x  = (const float*)d_in[0];
  const float* Ws = (const float*)d_in[1];
  const float* bs = (const float*)d_in[2];
  // d_in[3] = slice_bounds (int64): uniform equal slices, handled implicitly.
  float* out = (float*)d_out;

  unsigned short* whi = (unsigned short*)d_ws;
  unsigned short* wlo = whi + W_ELEMS;          // 2 * 33.5 MB in d_ws

  // 1) fp32 -> swizzled bf16 hi/lo weight fragments (one-time per call).
  int threads1 = (int)(W_ELEMS / 16);           // 1,048,576 threads
  weight_split_kernel<<<threads1 / 256, 256, 0, stream>>>(Ws, whi, wlo);

  // 2) fused 4-layer grouped GEMM over all models.
  int blocks = (N_MODELS * ROWS_PER_MODEL) / 64; // 4096 blocks, 64 rows each
  size_t shmem = 4 * 16 * N_FEAT * sizeof(float); // 64 KB
  ensemble_mlp_kernel<<<blocks, 128, shmem, stream>>>(x, bs, whi, wlo, out);
}
